// GCN4GC_1_81243601371607
// MI455X (gfx1250) — compile-verified, matrix operand order checked by reading, not run on hardware
//
#include <hip/hip_runtime.h>
#include <hip/hip_bf16.h>

// ---------------------------------------------------------------------------
// GCN (3-layer GCNConv + BN + ELU) -> mean-pool -> MLP head, for MI455X.
// Dominant cost: edge scatter (~2.5 GB irregular traffic) -> fp32 HW atomics.
// GEMMs use v_wmma_f32_16x16x32_bf16 with fragment-order pre-packed weights:
// inner loop = 2x ds_load_b128 (A) + 2x global_load_b128 (B) + v_wmma.
// ---------------------------------------------------------------------------

typedef __attribute__((ext_vector_type(16))) __bf16 v16bf;
typedef __attribute__((ext_vector_type(8)))  float  v8f;

#define DD 128          // hidden dim (reference D)
#define BN_EPS 1e-5f

__device__ __forceinline__ void atomAddF(float* p, float v) {
    __hip_atomic_fetch_add(p, v, __ATOMIC_RELAXED, __HIP_MEMORY_SCOPE_AGENT);
}

// -------------------------------- utility ----------------------------------
__global__ void k_zero_f32(float* __restrict__ p, long long n) {
    long long t = (long long)blockIdx.x * blockDim.x + threadIdx.x;
    if (t < n) p[t] = 0.0f;
}

// Pack a [D x D] f32 weight matrix (one or more layers) into WMMA B-fragment
// order: frag f = (colTile*4 + kstep)*32 + lane holds that lane's 16 bf16
// elements contiguously (32 B aligned) -> single v16bf load in the GEMM.
// Element e of lane maps to K = kstep*32 + e + (e>=8 ? 8 : 0) + 8*(lane>>4),
// column c = colTile*16 + (lane&15)   (ISA 7.12.2 16-bit layout, B symmetric).
__global__ void k_pack_wmma_b(const float* __restrict__ W, __bf16* __restrict__ Bp,
                              long long total) {              // total = layers*D*D
    long long t = (long long)blockIdx.x * blockDim.x + threadIdx.x;
    if (t >= total) return;
    int e     = (int)(t & 15);
    int lane  = (int)((t >> 4) & 31);
    int kstep = (int)((t >> 9) & 3);
    int ct    = (int)((t >> 11) & 7);
    long long layer = t >> 14;                                 // D*D = 16384
    int half = lane >> 4, l = lane & 15;
    int K = kstep * 32 + e + ((e >= 8) ? 8 : 0) + half * 8;
    int c = ct * 16 + l;
    Bp[t] = (__bf16)W[layer * (DD * DD) + (long long)K * DD + c];
}

// ------------------------------ degree / norm ------------------------------
__global__ void k_degree(const int* __restrict__ dst, float* __restrict__ deg, int E) {
    int t = blockIdx.x * blockDim.x + threadIdx.x;
    if (t < E) atomAddF(&deg[dst[t]], 1.0f);
}

__global__ void k_dis(float* __restrict__ deg_to_dis, int N) {
    int t = blockIdx.x * blockDim.x + threadIdx.x;
    if (t < N) deg_to_dis[t] = rsqrtf(deg_to_dis[t] + 1.0f);   // deg^{-1/2} incl self loop
}

__global__ void k_edge_norm(const int* __restrict__ src, const int* __restrict__ dst,
                            const float* __restrict__ dis, float* __restrict__ normE, int E) {
    int t = blockIdx.x * blockDim.x + threadIdx.x;
    if (t < E) normE[t] = dis[src[t]] * dis[dst[t]];
}

// ------------------------------ WMMA GEMM ----------------------------------
// C[M x 128] = A[M x 128](f32) * B(fragment-packed bf16, 128x128), M % 16 == 0.
// One block = 16 rows, 8 waves; wave w -> cols [16w, 16w+16).
// Optional fused epilogue: bias add + ReLU (for lin1).
__global__ void k_gemm_wmma(const float* __restrict__ A, const __bf16* __restrict__ Bpack,
                            float* __restrict__ C, int M,
                            const float* __restrict__ bias, int relu) {
    __shared__ __bf16 As[16 * DD];                 // 4 KB bf16 A tile
    const int tid     = threadIdx.x;
    const int rowBase = blockIdx.x * 16;

    // Stage + convert the 16x128 f32 A tile into LDS as bf16 (coalesced).
    for (int i = tid; i < 16 * DD; i += 256) {
        int r = i >> 7, c = i & (DD - 1);
        As[i] = (__bf16)A[(long long)(rowBase + r) * DD + c];
    }
    __syncthreads();

    const int wave = tid >> 5;
    const int lane = tid & 31;
    const int half = lane >> 4;                    // ISA half-wave group
    const int l    = lane & 15;
    const v16bf* Bp = (const v16bf*)Bpack;

    v8f acc = {};
    #pragma unroll
    for (int kk = 0; kk < 4; ++kk) {               // K = 128 in steps of 32
        const int k0 = kk * 32;
        v16bf a;
        #pragma unroll
        for (int e = 0; e < 16; ++e) {
            // A 16x32 bf16 layout: contiguous K runs of 8 -> ds_load_b128 x2.
            const int K = k0 + e + ((e >= 8) ? 8 : 0) + half * 8;
            a[e] = As[l * DD + K];
        }
        v16bf b = Bp[(wave * 4 + kk) * 32 + lane]; // 32 B vector load (b128 x2)
        acc = __builtin_amdgcn_wmma_f32_16x16x32_bf16(
            /*neg_a=*/false, a, /*neg_b=*/false, b,
            /*c_mod=*/(short)0, acc, /*reuse_a=*/false, /*reuse_b=*/false);
    }

    #pragma unroll
    for (int r = 0; r < 8; ++r) {                  // C/D: VGPR r -> M = r + 8*half
        const int m = rowBase + r + half * 8;
        const int c = wave * 16 + l;
        float v = acc[r];
        if (bias) v += bias[c];
        if (relu) v = v > 0.0f ? v : 0.0f;
        C[(long long)m * DD + c] = v;
    }
}

// ------------------------------ edge scatter -------------------------------
// agg[dst] += h[src] * norm[e];  32 threads per edge, float4 row reads.
__global__ void k_scatter(const float* __restrict__ H, const float* __restrict__ normE,
                          const int* __restrict__ src, const int* __restrict__ dst,
                          float* __restrict__ Agg, int E) {
    long long t = (long long)blockIdx.x * blockDim.x + threadIdx.x;
    int e = (int)(t >> 5);
    if (e >= E) return;
    int q = (int)(t & 31);                         // feature quad 0..31
    int s = src[e], d = dst[e];
    float w = normE[e];
    float4 hv = ((const float4*)(H + (long long)s * DD))[q];
    float* a = Agg + (long long)d * DD + q * 4;
    atomAddF(a + 0, hv.x * w);
    atomAddF(a + 1, hv.y * w);
    atomAddF(a + 2, hv.z * w);
    atomAddF(a + 3, hv.w * w);
}

// ---------------------- combine: GCN out + ELU (in place) ------------------
__global__ void k_combine_elu(float* __restrict__ Agg, const float* __restrict__ H,
                              const float* __restrict__ dis, const float* __restrict__ bvec,
                              long long ND) {
    long long t = (long long)blockIdx.x * blockDim.x + threadIdx.x;
    if (t >= ND) return;
    int n = (int)(t >> 7);
    int d = (int)(t & (DD - 1));
    float sn = dis[n]; sn *= sn;                   // self-loop weight dis^2
    float y  = Agg[t] + sn * H[t] + bvec[d];
    Agg[t]   = y > 0.0f ? y : expm1f(y);           // ELU(alpha=1)
}

// ----------------------- BN: column sum / sumsq ----------------------------
__global__ void k_col_reduce(const float* __restrict__ Y, float* __restrict__ sums,
                             int N) {
    const int d  = threadIdx.x;                    // 128 threads = one per column
    const int r0 = blockIdx.x * 128;
    float s = 0.0f, s2 = 0.0f;
    for (int r = 0; r < 128; ++r) {
        int row = r0 + r;
        if (row < N) {
            float v = Y[(long long)row * DD + d];
            s += v; s2 += v * v;
        }
    }
    atomAddF(&sums[d], s);
    atomAddF(&sums[DD + d], s2);
}

__global__ void k_bn_apply(const float* __restrict__ Y, float* __restrict__ X,
                           const float* __restrict__ sums,
                           const float* __restrict__ gamma, const float* __restrict__ beta,
                           int N, long long ND) {
    long long t = (long long)blockIdx.x * blockDim.x + threadIdx.x;
    if (t >= ND) return;
    int d = (int)(t & (DD - 1));
    float invN = 1.0f / (float)N;
    float mu   = sums[d] * invN;
    float var  = sums[DD + d] * invN - mu * mu;
    float inv  = rsqrtf(var + BN_EPS);
    X[t] = (Y[t] - mu) * inv * gamma[d] + beta[d];
}

// ------------------------------ global mean pool ---------------------------
__global__ void k_counts(const int* __restrict__ batch, float* __restrict__ counts, int N) {
    int t = blockIdx.x * blockDim.x + threadIdx.x;
    if (t < N) atomAddF(&counts[batch[t]], 1.0f);
}

__global__ void k_pool_sum(const float* __restrict__ X, const int* __restrict__ batch,
                           float* __restrict__ pooled, long long ND) {
    long long t = (long long)blockIdx.x * blockDim.x + threadIdx.x;
    if (t >= ND) return;
    int n = (int)(t >> 7);
    int d = (int)(t & (DD - 1));
    atomAddF(&pooled[(long long)batch[n] * DD + d], X[t]);
}

__global__ void k_pool_div(float* __restrict__ pooled, const float* __restrict__ counts,
                           int GD) {
    int t = blockIdx.x * blockDim.x + threadIdx.x;
    if (t < GD) pooled[t] /= fmaxf(counts[t >> 7], 1.0f);
}

// ------------------------- head: lin2 + log_softmax ------------------------
__global__ void k_head(const float* __restrict__ H1, const float* __restrict__ w2,
                       const float* __restrict__ b2, float* __restrict__ out,
                       int G, int C) {
    int g = blockIdx.x * blockDim.x + threadIdx.x;
    if (g >= G) return;
    float z[8];
    for (int c = 0; c < C; ++c) {
        float acc = b2[c];
        for (int d = 0; d < DD; ++d) acc += H1[(long long)g * DD + d] * w2[d * C + c];
        z[c] = acc;
    }
    float m = z[0];
    for (int c = 1; c < C; ++c) m = fmaxf(m, z[c]);
    float lse = 0.0f;
    for (int c = 0; c < C; ++c) lse += expf(z[c] - m);
    lse = m + logf(lse);
    for (int c = 0; c < C; ++c) out[(long long)g * C + c] = z[c] - lse;
}

// ---------------------------------------------------------------------------
static inline size_t alignUp(size_t v, size_t a) { return (v + a - 1) & ~(a - 1); }

extern "C" void kernel_launch(void* const* d_in, const int* in_sizes, int n_in,
                              void* d_out, int out_size, void* d_ws, size_t ws_size,
                              hipStream_t stream) {
    // Inputs (setup_inputs order):
    const float* x_in   = (const float*)d_in[0];   // [N,D]
    const float* W      = (const float*)d_in[1];   // [L,D,D]
    const float* bvec   = (const float*)d_in[2];   // [L,D]
    const float* gamma  = (const float*)d_in[3];   // [L,D]
    const float* beta   = (const float*)d_in[4];   // [L,D]
    const float* lin1_w = (const float*)d_in[5];   // [D,D]
    const float* lin1_b = (const float*)d_in[6];   // [D]
    const float* lin2_w = (const float*)d_in[7];   // [D,C]
    const float* lin2_b = (const float*)d_in[8];   // [C]
    const int*   eidx   = (const int*)d_in[9];     // [2,E]
    const int*   batch  = (const int*)d_in[10];    // [N]

    const int D = in_sizes[6];                     // 128
    const int N = in_sizes[10];                    // 50000 (multiple of 16)
    const int E = in_sizes[9] / 2;                 // 800000
    const int L = in_sizes[2] / D;                 // 3
    const int C = in_sizes[8];                     // 2
    const int G = out_size / C;                    // 128
    const long long ND = (long long)N * D;
    const int* src = eidx;
    const int* dst = eidx + E;

    // Workspace carve-up.
    char*  base = (char*)d_ws;
    size_t off  = 0;
    auto take = [&](size_t bytes) { size_t o = off; off = alignUp(off + bytes, 256); return (void*)(base + o); };
    float*  X      = (float*)take(sizeof(float) * ND);
    float*  H      = (float*)take(sizeof(float) * ND);
    float*  Agg    = (float*)take(sizeof(float) * ND);
    float*  dis    = (float*)take(sizeof(float) * N);
    float*  normE  = (float*)take(sizeof(float) * E);
    float*  sums   = (float*)take(sizeof(float) * 2 * D);
    float*  pooled = (float*)take(sizeof(float) * (size_t)G * D);
    float*  counts = (float*)take(sizeof(float) * G);
    float*  H1     = (float*)take(sizeof(float) * (size_t)G * D);
    __bf16* Wbf    = (__bf16*)take(sizeof(__bf16) * (size_t)L * D * D);  // packed frags
    __bf16* L1bf   = (__bf16*)take(sizeof(__bf16) * (size_t)D * D);     // packed frags
    (void)ws_size; (void)n_in;

    const int T = 256;
    auto blocks = [&](long long n) { return (unsigned)((n + T - 1) / T); };

    // --- one-time (per launch) prep ---
    k_zero_f32<<<blocks(N), T, 0, stream>>>(dis, N);
    k_zero_f32<<<blocks((long long)G * D), T, 0, stream>>>(pooled, (long long)G * D);
    k_zero_f32<<<blocks(G), T, 0, stream>>>(counts, G);
    k_degree<<<blocks(E), T, 0, stream>>>(dst, dis, E);
    k_dis<<<blocks(N), T, 0, stream>>>(dis, N);
    k_edge_norm<<<blocks(E), T, 0, stream>>>(src, dst, dis, normE, E);
    k_pack_wmma_b<<<blocks((long long)L * D * D), T, 0, stream>>>(W, Wbf, (long long)L * D * D);
    k_pack_wmma_b<<<blocks((long long)D * D), T, 0, stream>>>(lin1_w, L1bf, (long long)D * D);

    // --- L GCN layers ---
    const float* Xin = x_in;
    for (int i = 0; i < L; ++i) {
        k_zero_f32<<<blocks(ND), T, 0, stream>>>(Agg, ND);
        k_zero_f32<<<blocks(2 * D), T, 0, stream>>>(sums, 2 * D);
        // h = Xin @ W[i]   (WMMA bf16 -> f32)
        k_gemm_wmma<<<N / 16, 256, 0, stream>>>(Xin, Wbf + (size_t)i * D * D, H, N, nullptr, 0);
        // agg[dst] += h[src] * norm
        k_scatter<<<blocks((long long)E * 32), T, 0, stream>>>(H, normE, src, dst, Agg, E);
        // agg = elu(agg + dis^2 * h + b[i])
        k_combine_elu<<<blocks(ND), T, 0, stream>>>(Agg, H, dis, bvec + (size_t)i * D, ND);
        // BatchNorm (training-mode batch stats)
        k_col_reduce<<<(N + 127) / 128, 128, 0, stream>>>(Agg, sums, N);
        k_bn_apply<<<blocks(ND), T, 0, stream>>>(Agg, X, sums,
                                                 gamma + (size_t)i * D, beta + (size_t)i * D, N, ND);
        Xin = X;
    }

    // --- global mean pool ---
    k_counts<<<blocks(N), T, 0, stream>>>(batch, counts, N);
    k_pool_sum<<<blocks(ND), T, 0, stream>>>(X, batch, pooled, ND);
    k_pool_div<<<blocks((long long)G * D), T, 0, stream>>>(pooled, counts, G * D);

    // --- MLP head: h1 = relu(pooled @ lin1_w + lin1_b) via WMMA, then lin2 + log_softmax
    k_gemm_wmma<<<G / 16, 256, 0, stream>>>(pooled, L1bf, H1, G, lin1_b, 1);
    k_head<<<blocks(G), T, 0, stream>>>(H1, lin2_w, lin2_b, (float*)d_out, G, C);
}
